// QDGRU_65274912964831
// MI455X (gfx1250) — matrix operand order repaired and checked by reading, not compiled, not run on hardware
//
#include <hip/hip_runtime.h>
#include <math.h>

#define SEQ   168
#define SEQP  192       // K padded to 6 x 32
#define LDA   200       // LDS row stride (halves): 400B rows keep 16B alignment
#define NF    8
#define PRED  24
#define KER   25
#define CTXN  32
#define HID   8
#define NQ    4
#define PI_F  3.14159265358979f

typedef __attribute__((ext_vector_type(16))) _Float16 v16h;
typedef __attribute__((ext_vector_type(8)))  _Float16 v8h;
typedef __attribute__((ext_vector_type(8)))  float    v8f;

struct __align__(16) Smem {
  _Float16 x16[2][16][LDA];   // per-wave padded input tile (f16)
  _Float16 tr16[2][16][LDA];  // per-wave trend tile (f16, K-padded w/ zeros)
  _Float16 se16[2][16][LDA];  // per-wave seasonal tile
  _Float16 wtr[32][LDA];      // trend_w  as f16, zero-padded [32][192]
  _Float16 wse[32][LDA];      // seasonal_w
  _Float16 wta[32][LDA];      // ta_w
  float ctx[16][NF][CTXN];    // b,f,c (+ta_b)
  float wsoft[NF][KER];       // softmaxed decomp weights
  float gin[16][HID];
  float h0s[16][HID];
  float rz[16][NQ];
  float nin[16][NQ];
  float rg[16][HID];
  float zg[16][HID];
  float hn[16][HID];
  float modl[16][PRED];
};

// A fragment: 16x32 f16, ISA layout. lane group g = lane>>4, m = lane&15.
// element e<8 : K = kb + 8g + e ; e>=8 : K = kb + 16 + 8g + (e-8)
// Both 8-half runs are 16B aligned -> two ds_load_b128.
__device__ inline v16h load_A_lds(const _Float16* base, int lane, int kb) {
  int m = lane & 15, g = lane >> 4;
  const _Float16* p = base + m * LDA + kb + 8 * g;
  v8h lo = *(const v8h*)p;
  v8h hi = *(const v8h*)(p + 16);
  return __builtin_shufflevector(lo, hi, 0, 1, 2, 3, 4, 5, 6, 7,
                                         8, 9, 10, 11, 12, 13, 14, 15);
}

// B fragment: 32x16 f16 (K x N) from zero-padded LDS weights [32][LDA].
// lane group g: e<8: K = kb + 8g + e ; e>=8: K = kb + 16 + 8g + (e-8) ; N = (lane&15) + 16*ntile  (two b128 loads)
__device__ inline v16h load_B_lds(const _Float16* w16, int lane, int kb, int nt) {
  int n = (lane & 15) + 16 * nt, g = lane >> 4;
  const _Float16* p = w16 + n * LDA + kb + 8 * g;
  v8h lo = *(const v8h*)p;
  v8h hi = *(const v8h*)(p + 16);
  return __builtin_shufflevector(lo, hi, 0, 1, 2, 3, 4, 5, 6, 7,
                                         8, 9, 10, 11, 12, 13, 14, 15);
}

// 4-qubit statevector sim of the reference VQC (2 layers: RY(in), Rot(w), CNOT ring)
__device__ inline void vqc4(const float* in4, const float* w, float* exps) {
  float sr[16], si[16];
#pragma unroll
  for (int i = 0; i < 16; ++i) { sr[i] = 0.f; si[i] = 0.f; }
  sr[0] = 1.f;
#pragma unroll
  for (int l = 0; l < 2; ++l) {
#pragma unroll
    for (int q = 0; q < 4; ++q) {         // RY(inputs[q])
      float c = cosf(0.5f * in4[q]);
      float s = sinf(0.5f * in4[q]);
      int st = 8 >> q;
#pragma unroll
      for (int i = 0; i < 16; ++i) {
        if (i & st) continue;
        int j = i | st;
        float ar = sr[i], ai = si[i], br = sr[j], bi = si[j];
        sr[i] = c * ar - s * br;  si[i] = c * ai - s * bi;
        sr[j] = s * ar + c * br;  si[j] = s * ai + c * bi;
      }
    }
#pragma unroll
    for (int q = 0; q < 4; ++q) {         // Rot(phi,theta,omega)
      const float* wq = w + (l * 4 + q) * 3;
      float phi = wq[0], th = wq[1], om = wq[2];
      float ch = cosf(0.5f * th), sh = sinf(0.5f * th);
      float ap = -0.5f * (phi + om), am = 0.5f * (phi - om);
      float epr = cosf(ap), epi = sinf(ap);
      float emr = cosf(am), emi = sinf(am);
      float u00r = epr * ch, u00i = epi * ch;
      float u01r = -emr * sh, u01i = -emi * sh;
      float u10r =  emr * sh, u10i = -emi * sh;
      float u11r =  epr * ch, u11i = -epi * ch;
      int st = 8 >> q;
#pragma unroll
      for (int i = 0; i < 16; ++i) {
        if (i & st) continue;
        int j = i | st;
        float ar = sr[i], ai = si[i], br = sr[j], bi = si[j];
        sr[i] = u00r * ar - u00i * ai + u01r * br - u01i * bi;
        si[i] = u00r * ai + u00i * ar + u01r * bi + u01i * br;
        sr[j] = u10r * ar - u10i * ai + u11r * br - u11i * bi;
        si[j] = u10r * ai + u10i * ar + u11r * bi + u11i * br;
      }
    }
#pragma unroll
    for (int q = 0; q < 4; ++q) {         // CNOT(q, (q+1)%4)
      int cs = 8 >> q;
      int ts = 8 >> ((q + 1) & 3);
#pragma unroll
      for (int i = 0; i < 16; ++i) {
        if (!(i & cs) || (i & ts)) continue;
        int j = i | ts;
        float tr = sr[i], ti = si[i];
        sr[i] = sr[j]; si[i] = si[j];
        sr[j] = tr;    si[j] = ti;
      }
    }
  }
#pragma unroll
  for (int q = 0; q < 4; ++q) {           // <Z_q>
    int st = 8 >> q;
    float e = 0.f;
#pragma unroll
    for (int i = 0; i < 16; ++i) {
      float p = sr[i] * sr[i] + si[i] * si[i];
      e += (i & st) ? -p : p;
    }
    exps[q] = e;
  }
}

__launch_bounds__(64)
__global__ void qdgru_fused(
    const float* __restrict__ x, const float* __restrict__ decomp_w,
    const float* __restrict__ trend_w, const float* __restrict__ trend_b,
    const float* __restrict__ seasonal_w, const float* __restrict__ seasonal_b,
    const float* __restrict__ h0_w, const float* __restrict__ h0_b,
    const float* __restrict__ ta_w, const float* __restrict__ ta_b,
    const float* __restrict__ sc_w, const float* __restrict__ sc_b,
    const float* __restrict__ crz_w, const float* __restrict__ crz_b,
    const float* __restrict__ cn_w, const float* __restrict__ cn_b,
    const float* __restrict__ vqc_r, const float* __restrict__ vqc_u,
    const float* __restrict__ vqc_c,
    const float* __restrict__ expand_w, const float* __restrict__ expand_b,
    const float* __restrict__ mod_w, const float* __restrict__ mod_b,
    const float* __restrict__ fusion_alpha,
    float* __restrict__ out, int Btot) {
  __shared__ Smem S;
  const int t    = threadIdx.x;
  const int wv   = t >> 5;      // wave 0..1
  const int lane = t & 31;
  const int b0   = blockIdx.x * 16;

  // ---- prologue: softmax of decomp weights + stage f16 weights (zero-padded) ----
  if (t < NF) {
    int f = t;
    float mx = -1e30f;
    for (int k = 0; k < KER; ++k) mx = fmaxf(mx, decomp_w[f * KER + k]);
    float sm = 0.f;
    for (int k = 0; k < KER; ++k) sm += expf(decomp_w[f * KER + k] - mx);
    float inv = 1.f / sm;
    for (int k = 0; k < KER; ++k)
      S.wsoft[f][k] = expf(decomp_w[f * KER + k] - mx) * inv;
  }
  {
    const float* srcs[3] = {trend_w, seasonal_w, ta_w};
    _Float16*    dsts[3] = {&S.wtr[0][0], &S.wse[0][0], &S.wta[0][0]};
    const int    rows[3] = {PRED, PRED, CTXN};
#pragma unroll
    for (int mi = 0; mi < 3; ++mi) {
      const float* w = srcs[mi];
      _Float16* dst  = dsts[mi];
      int r = rows[mi];
      for (int idx = t; idx < 32 * SEQP; idx += 64) {
        int n = idx / SEQP, k = idx % SEQP;
        float v = (n < r && k < SEQ) ? w[n * SEQ + k] : 0.f;
        dst[n * LDA + k] = (_Float16)v;
      }
    }
  }
  __syncthreads();

  // ---- phase 1: conv + WMMA GEMMs over seq. wave wv owns features {wv, wv+2, wv+4, wv+6} ----
  v8f accT[4][2], accS[4][2];
#pragma unroll
  for (int fi = 0; fi < 4; ++fi)
#pragma unroll
    for (int nt = 0; nt < 2; ++nt) { accT[fi][nt] = (v8f){}; accS[fi][nt] = (v8f){}; }

#pragma unroll
  for (int fi = 0; fi < 4; ++fi) {
    const int f = wv + 2 * fi;
    // edge-padded input tile -> f16 LDS
    for (int idx = lane; idx < 16 * SEQP; idx += 32) {
      int m = idx / SEQP, j = idx % SEQP;
      int bb = b0 + m; if (bb >= Btot) bb = Btot - 1;
      int s = j - 12; s = s < 0 ? 0 : (s > SEQ - 1 ? SEQ - 1 : s);
      S.x16[wv][m][j] = (_Float16)x[((size_t)bb * SEQ + s) * NF + f];
    }
    __syncthreads();
    // depthwise conv (K=25): each lane computes 8 consecutive outputs per chunk.
    // Window = 32 halves (4 x b128 aligned loads); store results as v8h (b128).
    {
      float wk[KER];
#pragma unroll
      for (int k = 0; k < KER; ++k) wk[k] = S.wsoft[f][k];
      for (int idx = lane; idx < 16 * 21; idx += 32) {
        int m = idx / 21, s0 = (idx % 21) * 8;
        const _Float16* row = &S.x16[wv][m][0];
        v8h w0 = *(const v8h*)(row + s0);
        v8h w1 = *(const v8h*)(row + s0 + 8);
        v8h w2 = *(const v8h*)(row + s0 + 16);
        v8h w3 = *(const v8h*)(row + s0 + 24);
        float win[32];
#pragma unroll
        for (int e = 0; e < 8; ++e) {
          win[e]      = (float)w0[e];
          win[8 + e]  = (float)w1[e];
          win[16 + e] = (float)w2[e];
          win[24 + e] = (float)w3[e];
        }
        v8h tv, sv;
#pragma unroll
        for (int j = 0; j < 8; ++j) {
          float acc = 0.f;
#pragma unroll
          for (int k = 0; k < KER; ++k) acc += wk[k] * win[j + k];
          tv[j] = (_Float16)acc;
          sv[j] = (_Float16)(win[j + 12] - acc);
        }
        *(v8h*)(&S.tr16[wv][m][s0]) = tv;
        *(v8h*)(&S.se16[wv][m][s0]) = sv;
      }
    }
    // zero K-padding (s = 168..191), vectorized
    for (int idx = lane; idx < 16 * 3; idx += 32) {
      int m = idx / 3, s0 = SEQ + (idx % 3) * 8;
      v8h z = {};
      *(v8h*)(&S.tr16[wv][m][s0]) = z;
      *(v8h*)(&S.se16[wv][m][s0]) = z;
    }
    __syncthreads();
    // three GEMMs: [16 x 192] x [192 x {24,24,32}] via v_wmma_f32_16x16x32_f16
    v8f accC[2] = {(v8f){}, (v8f){}};
#pragma unroll
    for (int kc = 0; kc < 6; ++kc) {
      const int kb = kc * 32;
      v16h aT = load_A_lds(&S.tr16[wv][0][0], lane, kb);
      v16h aS = load_A_lds(&S.se16[wv][0][0], lane, kb);
#pragma unroll
      for (int nt = 0; nt < 2; ++nt) {
        v16h bT = load_B_lds(&S.wtr[0][0], lane, kb, nt);
        accT[fi][nt] = __builtin_amdgcn_wmma_f32_16x16x32_f16(
            false, aT, false, bT, (short)0, accT[fi][nt], false, false);
        v16h bS = load_B_lds(&S.wse[0][0], lane, kb, nt);
        accS[fi][nt] = __builtin_amdgcn_wmma_f32_16x16x32_f16(
            false, aS, false, bS, (short)0, accS[fi][nt], false, false);
        v16h bC = load_B_lds(&S.wta[0][0], lane, kb, nt);
        accC[nt] = __builtin_amdgcn_wmma_f32_16x16x32_f16(
            false, aS, false, bC, (short)0, accC[nt], false, false);
      }
    }
    // ctx -> LDS (+ta_b)
#pragma unroll
    for (int nt = 0; nt < 2; ++nt) {
      int c = (lane & 15) + 16 * nt;
#pragma unroll
      for (int r = 0; r < 8; ++r) {
        int m = r + 8 * (lane >> 4);
        S.ctx[m][f][c] = accC[nt][r] + ta_b[c];
      }
    }
    __syncthreads();
  }

  // ---- phase 2: scalar GRU / VQC chain (16 batch elems, 64 threads) ----
  if (t < 16) {
    const int lagidx[11] = {167, 166, 165, 164, 163, 162, 145, 144, 143, 1, 0};
    int bb = b0 + t; if (bb >= Btot) bb = Btot - 1;
    float lv[11];
#pragma unroll
    for (int j = 0; j < 11; ++j)
      lv[j] = x[((size_t)bb * SEQ + lagidx[j]) * NF + (NF - 1)];
#pragma unroll
    for (int o = 0; o < HID; ++o) {
      float s = h0_b[o];
#pragma unroll
      for (int j = 0; j < 11; ++j) s += lv[j] * h0_w[o * 11 + j];
      S.h0s[t][o] = tanhf(s);
    }
  }
  for (int task = t; task < 16 * HID; task += 64) {   // gru_in
    int b = task >> 3, o = task & 7;
    float s = sc_b[o];
    for (int f = 0; f < NF; ++f)
#pragma unroll
      for (int c = 0; c < CTXN; ++c)
        s += S.ctx[b][f][c] * sc_w[o * (NF * CTXN) + f * CTXN + c];
    S.gin[b][o] = tanhf(s);
  }
  __syncthreads();
  if (t < 16) {                                        // rz_in
#pragma unroll
    for (int q = 0; q < NQ; ++q) {
      float s = crz_b[q];
#pragma unroll
      for (int j = 0; j < HID; ++j) s += S.gin[t][j] * crz_w[q * 16 + j];
#pragma unroll
      for (int j = 0; j < HID; ++j) s += S.h0s[t][j] * crz_w[q * 16 + 8 + j];
      S.rz[t][q] = tanhf(s) * PI_F;
    }
  }
  __syncthreads();
  if (t < 32) {                                        // r and z VQCs
    int b = t & 15, kind = t >> 4;
    float in4[4] = {S.rz[b][0], S.rz[b][1], S.rz[b][2], S.rz[b][3]};
    float e[4];
    vqc4(in4, kind ? vqc_u : vqc_r, e);
    float (*dst)[HID] = kind ? S.zg : S.rg;
#pragma unroll
    for (int o = 0; o < HID; ++o) {
      float v = expand_b[o];
#pragma unroll
      for (int q = 0; q < NQ; ++q) v += e[q] * expand_w[o * NQ + q];
      dst[b][o] = 1.f / (1.f + expf(-v));
    }
  }
  __syncthreads();
  if (t < 16) {                                        // n_in
#pragma unroll
    for (int q = 0; q < NQ; ++q) {
      float s = cn_b[q];
#pragma unroll
      for (int j = 0; j < HID; ++j) s += S.rg[t][j] * S.h0s[t][j] * cn_w[q * 16 + j];
#pragma unroll
      for (int j = 0; j < HID; ++j) s += S.gin[t][j] * cn_w[q * 16 + 8 + j];
      S.nin[t][q] = tanhf(s) * PI_F;
    }
  }
  __syncthreads();
  if (t < 16) {                                        // n VQC + h_new
    float in4[4] = {S.nin[t][0], S.nin[t][1], S.nin[t][2], S.nin[t][3]};
    float e[4];
    vqc4(in4, vqc_c, e);
#pragma unroll
    for (int o = 0; o < HID; ++o) {
      float v = expand_b[o];
#pragma unroll
      for (int q = 0; q < NQ; ++q) v += e[q] * expand_w[o * NQ + q];
      float n = tanhf(v);
      float z = S.zg[t][o];
      S.hn[t][o] = (1.f - z) * n + z * S.h0s[t][o];
    }
  }
  __syncthreads();
  for (int task = t; task < 16 * PRED; task += 64) {   // modulation
    int b = task / PRED, p = task % PRED;
    float s = mod_b[p];
#pragma unroll
    for (int o = 0; o < HID; ++o) s += S.hn[b][o] * mod_w[p * HID + o];
    S.modl[b][p] = tanhf(s);
  }
  __syncthreads();

  // ---- phase 3: fused output from WMMA accumulators ----
  const float alpha = 1.f / (1.f + expf(-fusion_alpha[0]));
#pragma unroll
  for (int fi = 0; fi < 4; ++fi) {
    const int f = wv + 2 * fi;
#pragma unroll
    for (int nt = 0; nt < 2; ++nt) {
      int p = (lane & 15) + 16 * nt;
      if (p < PRED) {
#pragma unroll
        for (int r = 0; r < 8; ++r) {
          int m = r + 8 * (lane >> 4);
          if (b0 + m < Btot) {
            float tv = accT[fi][nt][r] + trend_b[p];
            float sv = (accS[fi][nt][r] + seasonal_b[p]) * (1.f + S.modl[m][p]);
            out[((size_t)(b0 + m) * PRED + p) * NF + f] = alpha * sv + (1.f - alpha) * tv;
          }
        }
      }
    }
  }
}

extern "C" void kernel_launch(void* const* d_in, const int* in_sizes, int n_in,
                              void* d_out, int out_size, void* d_ws, size_t ws_size,
                              hipStream_t stream) {
  const float* x         = (const float*)d_in[0];
  const float* decomp_w  = (const float*)d_in[1];
  const float* trend_w   = (const float*)d_in[2];
  const float* trend_b   = (const float*)d_in[3];
  const float* season_w  = (const float*)d_in[4];
  const float* season_b  = (const float*)d_in[5];
  const float* h0_w      = (const float*)d_in[6];
  const float* h0_b      = (const float*)d_in[7];
  const float* ta_w      = (const float*)d_in[8];
  const float* ta_b      = (const float*)d_in[9];
  const float* sc_w      = (const float*)d_in[10];
  const float* sc_b      = (const float*)d_in[11];
  const float* crz_w     = (const float*)d_in[12];
  const float* crz_b     = (const float*)d_in[13];
  const float* cn_w      = (const float*)d_in[14];
  const float* cn_b      = (const float*)d_in[15];
  const float* vqc_r     = (const float*)d_in[16];
  const float* vqc_u     = (const float*)d_in[17];
  const float* vqc_c     = (const float*)d_in[18];
  const float* expand_w  = (const float*)d_in[19];
  const float* expand_b  = (const float*)d_in[20];
  const float* mod_w     = (const float*)d_in[21];
  const float* mod_b     = (const float*)d_in[22];
  const float* fus_alpha = (const float*)d_in[23];

  int Btot = in_sizes[0] / (SEQ * NF);
  int blocks = (Btot + 15) / 16;
  qdgru_fused<<<blocks, 64, 0, stream>>>(
      x, decomp_w, trend_w, trend_b, season_w, season_b, h0_w, h0_b,
      ta_w, ta_b, sc_w, sc_b, crz_w, crz_b, cn_w, cn_b,
      vqc_r, vqc_u, vqc_c, expand_w, expand_b, mod_w, mod_b, fus_alpha,
      (float*)d_out, Btot);
}
